// DualAttention_3186865734514
// MI455X (gfx1250) — matrix operand order repaired and checked by reading, not compiled, not run on hardware
//
#include <hip/hip_runtime.h>
#include <hip/hip_bf16.h>
#include <stdint.h>

// Problem constants (match reference)
#define B_   4
#define C_   512
#define N_   4096
#define CQ_  64

typedef __attribute__((ext_vector_type(16))) __bf16 v16bf;
typedef __attribute__((ext_vector_type(8)))  float  v8f;

__device__ __forceinline__ uint16_t f2bf(float f) {
    union { float f; uint32_t u; } v; v.f = f;
    uint32_t u = v.u;
    u += 0x7FFFu + ((u >> 16) & 1u);   // round-to-nearest-even
    return (uint16_t)(u >> 16);
}

// ---------------------------------------------------------------------------
// WMMA fragment loaders (wave32, 16x16x32 bf16).
// A (16xK=32): lane&15 = M row; K pairs: k = 8*lane[4] + 16*(v>>2) + 2*(v&3)+s
// B (K=32x16): lane&15 = N col; k = 8*lane[4] + 16*(v>>2) + 2*(v&3)+s (same as A)
// B is loaded from the *transposed* source (rows = N-dim, ld = K-dim), so every
// lane reads two runs of 16 contiguous bytes.
// ---------------------------------------------------------------------------
__device__ __forceinline__ v16bf load_a16(const uint16_t* base, int row0, int ld, int k0) {
    int lane = threadIdx.x & 31;
    const uint16_t* p = base + (size_t)(row0 + (lane & 15)) * ld + k0 + ((lane >> 4) << 3);
    union { v16bf v; uint32_t u[8]; } f;
#pragma unroll
    for (int i = 0; i < 8; ++i)
        f.u[i] = *(const uint32_t*)(p + ((i >> 2) << 4) + ((i & 3) << 1));
    return f.v;
}

__device__ __forceinline__ v16bf load_b16(const uint16_t* baseT, int col0, int ld, int k0) {
    int lane = threadIdx.x & 31;
    const uint16_t* p = baseT + (size_t)(col0 + (lane & 15)) * ld + k0 + ((lane >> 4) << 3);
    union { v16bf v; uint32_t u[8]; } f;
#pragma unroll
    for (int i = 0; i < 8; ++i)
        f.u[i] = *(const uint32_t*)(p + ((i >> 2) << 4) + ((i & 3) << 1));
    return f.v;
}

__device__ __forceinline__ v8f wmma_bf16(v16bf a, v16bf b, v8f c) {
    return __builtin_amdgcn_wmma_f32_16x16x32_bf16(false, a, false, b, (short)0, c, false, false);
}

// ---------------------------------------------------------------------------
// 0) f32 -> bf16 casts
// ---------------------------------------------------------------------------
__global__ void k_cast_bf16(const float* __restrict__ src, uint16_t* __restrict__ dst, int n) {
    int i = blockIdx.x * blockDim.x + threadIdx.x;
    if (i < n) dst[i] = f2bf(src[i]);
}

// xt[b][n][c] = bf16(x[b][c][n])   (B*N*C threads, c fastest)
__global__ void k_x_transpose(const float* __restrict__ x, uint16_t* __restrict__ xt) {
    int i = blockIdx.x * blockDim.x + threadIdx.x;
    int c = i & (C_ - 1);
    int n = (i >> 9) & (N_ - 1);
    int b = i >> 21;
    xt[i] = f2bf(x[((size_t)b * C_ + c) * N_ + n]);
}

// ---------------------------------------------------------------------------
// 1) Projections: out[o,n] = sum_c W[o,c] x[c,n] + bias[o]
//    transposedOut=1 -> out is (N,O) per batch (qt/kt), else (O,N) (v)
// ---------------------------------------------------------------------------
__global__ void k_proj(const uint16_t* __restrict__ Wb, const float* __restrict__ bias,
                       const uint16_t* __restrict__ xt, int O,
                       uint16_t* __restrict__ out, int transposedOut) {
    int lane = threadIdx.x & 31;
    int wid  = blockIdx.x * (blockDim.x >> 5) + (threadIdx.x >> 5);
    int tilesPerB = (O >> 4) * (N_ >> 4);
    int b  = wid / tilesPerB;
    int t  = wid - b * tilesPerB;
    int o0 = (t / (N_ >> 4)) << 4;
    int n0 = (t % (N_ >> 4)) << 4;
    const uint16_t* xb = xt + (size_t)b * N_ * C_;
    v8f acc = {};
#pragma unroll 4
    for (int c0 = 0; c0 < C_; c0 += 32) {
        v16bf a  = load_a16(Wb, o0, C_, c0);
        v16bf bb = load_b16(xb, n0, C_, c0);
        acc = wmma_bf16(a, bb, acc);
    }
    int hi = lane >> 4, nn = lane & 15;
#pragma unroll
    for (int e = 0; e < 8; ++e) {
        int o = o0 + e + (hi << 3);
        int n = n0 + nn;
        float v = acc[e] + bias[o];
        if (transposedOut) out[((size_t)b * N_ + n) * O + o] = f2bf(v);
        else               out[((size_t)b * O + o) * N_ + n] = f2bf(v);
    }
}

// ---------------------------------------------------------------------------
// 2) PAM pass A: streaming online-softmax stats per row i (no N x N matrix)
// ---------------------------------------------------------------------------
__global__ void k_pam_stats(const uint16_t* __restrict__ qt, const uint16_t* __restrict__ kt,
                            float* __restrict__ rowmax, float* __restrict__ rowsum) {
    int lane = threadIdx.x & 31;
    int wid  = blockIdx.x * (blockDim.x >> 5) + (threadIdx.x >> 5);
    int b  = wid >> 8;               // 256 i-tiles per batch
    int i0 = (wid & 255) << 4;
    const uint16_t* qb = qt + (size_t)b * N_ * CQ_;
    const uint16_t* kb = kt + (size_t)b * N_ * CQ_;
    v16bf aq0 = load_a16(qb, i0, CQ_, 0);
    v16bf aq1 = load_a16(qb, i0, CQ_, 32);
    float mx[8], sm[8];
#pragma unroll
    for (int e = 0; e < 8; ++e) { mx[e] = -3.0e38f; sm[e] = 0.f; }
    for (int j0 = 0; j0 < N_; j0 += 16) {
        __builtin_prefetch(kb + (size_t)(j0 + 64) * CQ_, 0, 1);
        v16bf b0 = load_b16(kb, j0, CQ_, 0);
        v16bf b1 = load_b16(kb, j0, CQ_, 32);
        v8f acc = {};
        acc = wmma_bf16(aq0, b0, acc);
        acc = wmma_bf16(aq1, b1, acc);
#pragma unroll
        for (int e = 0; e < 8; ++e) {
            float l = acc[e];
            if (l <= mx[e]) {
                sm[e] += __expf(l - mx[e]);
            } else {
                sm[e] = sm[e] * __expf(mx[e] - l) + 1.0f;
                mx[e] = l;
            }
        }
    }
    // merge across the 16 lanes sharing the same row set (lane[4] preserved)
#pragma unroll
    for (int e = 0; e < 8; ++e) {
        float m = mx[e], s = sm[e];
#pragma unroll
        for (int off = 1; off < 16; off <<= 1) {
            float mo = __shfl_xor(m, off, 32);
            float so = __shfl_xor(s, off, 32);
            float mn = fmaxf(m, mo);
            s = s * __expf(m - mn) + so * __expf(mo - mn);
            m = mn;
        }
        mx[e] = m; sm[e] = s;
    }
    if ((lane & 15) == 0) {
        int hi = lane >> 4;
#pragma unroll
        for (int e = 0; e < 8; ++e) {
            int i = i0 + e + (hi << 3);
            rowmax[b * N_ + i] = mx[e];
            rowsum[b * N_ + i] = sm[e];
        }
    }
}

// ---------------------------------------------------------------------------
// 3) PAM pass B: recompute att tile (16x32) -> LDS, multiply with V, fuse
//    pam = gamma*pout + x.  8 waves: waves 0,1 produce, all 8 consume (64 c each).
// ---------------------------------------------------------------------------
__global__ void k_pam_apply(const uint16_t* __restrict__ qt, const uint16_t* __restrict__ kt,
                            const uint16_t* __restrict__ vb, const float* __restrict__ rowmax,
                            const float* __restrict__ rowsum, const float* __restrict__ x,
                            const float* __restrict__ gamma_pam,
                            float* __restrict__ pamf, uint16_t* __restrict__ pamb,
                            uint16_t* __restrict__ pamt) {
    __shared__ uint16_t satt[16 * 32];
    int lane = threadIdx.x & 31;
    int wave = threadIdx.x >> 5;
    int b  = blockIdx.x >> 8;
    int i0 = (blockIdx.x & 255) << 4;
    const uint16_t* qb  = qt + (size_t)b * N_ * CQ_;
    const uint16_t* kb  = kt + (size_t)b * N_ * CQ_;
    const uint16_t* vbb = vb + (size_t)b * C_ * N_;
    int hi = lane >> 4, nn = lane & 15;

    v16bf aq0 = {}, aq1 = {};
    float rm[8], rsi[8];
#pragma unroll
    for (int e = 0; e < 8; ++e) { rm[e] = 0.f; rsi[e] = 0.f; }
    if (wave < 2) {
        aq0 = load_a16(qb, i0, CQ_, 0);
        aq1 = load_a16(qb, i0, CQ_, 32);
#pragma unroll
        for (int e = 0; e < 8; ++e) {
            int i = i0 + e + (hi << 3);
            rm[e]  = rowmax[b * N_ + i];
            rsi[e] = 1.0f / rowsum[b * N_ + i];
        }
    }
    int c0w = wave << 6;                 // 64 channels per wave
    v8f acc[4] = { {}, {}, {}, {} };
    for (int j0 = 0; j0 < N_; j0 += 32) {
        if (wave < 2) {                  // wave-uniform: EXEC stays all-ones
            int jh = j0 + (wave << 4);
            v16bf b0 = load_b16(kb, jh, CQ_, 0);
            v16bf b1 = load_b16(kb, jh, CQ_, 32);
            v8f e8 = {};
            e8 = wmma_bf16(aq0, b0, e8);
            e8 = wmma_bf16(aq1, b1, e8);
#pragma unroll
            for (int e = 0; e < 8; ++e) {
                int m = e + (hi << 3);
                float p = __expf(e8[e] - rm[e]) * rsi[e];
                satt[m * 32 + (wave << 4) + nn] = f2bf(p);
            }
        }
        __syncthreads();
        v16bf af = load_a16(satt, 0, 32, 0);
#pragma unroll
        for (int t = 0; t < 4; ++t) {
            __builtin_prefetch(vbb + (size_t)(c0w + (t << 4) + nn) * N_ + j0 + 64, 0, 1);
            v16bf bf = load_b16(vbb, c0w + (t << 4), N_, j0);
            acc[t] = wmma_bf16(af, bf, acc[t]);
        }
        __syncthreads();
    }
    float gp = gamma_pam[0];
#pragma unroll
    for (int t = 0; t < 4; ++t) {
#pragma unroll
        for (int e = 0; e < 8; ++e) {
            int i = i0 + e + (hi << 3);
            int c = c0w + (t << 4) + nn;
            size_t idx = ((size_t)b * C_ + c) * N_ + i;
            float v = gp * acc[t][e] + x[idx];
            pamf[idx] = v;
            uint16_t h = f2bf(v);
            pamb[idx] = h;
            pamt[((size_t)b * N_ + i) * C_ + c] = h;
        }
    }
}

// ---------------------------------------------------------------------------
// 4) CAM energy: cen[c,d] = sum_n pf[c,n] pf[d,n]
// ---------------------------------------------------------------------------
__global__ void k_cen(const uint16_t* __restrict__ pamb, float* __restrict__ cen) {
    int lane = threadIdx.x & 31;
    int wid  = blockIdx.x * (blockDim.x >> 5) + (threadIdx.x >> 5);
    int b  = wid >> 10;                  // 32*32 tiles per batch
    int t  = wid & 1023;
    int c0 = (t >> 5) << 4;
    int d0 = (t & 31) << 4;
    const uint16_t* pb = pamb + (size_t)b * C_ * N_;
    v8f acc = {};
    for (int j0 = 0; j0 < N_; j0 += 32) {
        __builtin_prefetch(pb + (size_t)(c0 + (lane & 15)) * N_ + j0 + 128, 0, 1);
        v16bf a  = load_a16(pb, c0, N_, j0);
        v16bf bb = load_b16(pb, d0, N_, j0);
        acc = wmma_bf16(a, bb, acc);
    }
    int hi = lane >> 4, nn = lane & 15;
#pragma unroll
    for (int e = 0; e < 8; ++e) {
        int c = c0 + e + (hi << 3);
        cen[((size_t)b * C_ + c) * C_ + d0 + nn] = acc[e];
    }
}

// ---------------------------------------------------------------------------
// 5) CAM softmax: softmax(max - cen) == exp(min - cen) / sum exp(min - cen)
//    One wave per 512-wide row.
// ---------------------------------------------------------------------------
__global__ void k_catt(const float* __restrict__ cen, uint16_t* __restrict__ catt) {
    int lane = threadIdx.x & 31;
    int wid  = blockIdx.x * (blockDim.x >> 5) + (threadIdx.x >> 5);  // row over B*C
    const float* row = cen + (size_t)wid * C_;
    float v[16];
    float mn = 3.0e38f;
#pragma unroll
    for (int k = 0; k < 16; ++k) {
        v[k] = row[lane + (k << 5)];
        mn = fminf(mn, v[k]);
    }
#pragma unroll
    for (int off = 1; off < 32; off <<= 1)
        mn = fminf(mn, __shfl_xor(mn, off, 32));
    float s = 0.f;
#pragma unroll
    for (int k = 0; k < 16; ++k) { v[k] = __expf(mn - v[k]); s += v[k]; }
#pragma unroll
    for (int off = 1; off < 32; off <<= 1)
        s += __shfl_xor(s, off, 32);
    float inv = 1.0f / s;
    uint16_t* o = catt + (size_t)wid * C_;
#pragma unroll
    for (int k = 0; k < 16; ++k) o[lane + (k << 5)] = f2bf(v[k] * inv);
}

// ---------------------------------------------------------------------------
// 6) CAM apply: out = gamma_cam * (catt @ pf) + pam
// ---------------------------------------------------------------------------
__global__ void k_cam_apply(const uint16_t* __restrict__ catt, const uint16_t* __restrict__ pamt,
                            const float* __restrict__ pamf, const float* __restrict__ gamma_cam,
                            float* __restrict__ out) {
    int lane = threadIdx.x & 31;
    int wid  = blockIdx.x * (blockDim.x >> 5) + (threadIdx.x >> 5);
    int b  = wid >> 13;                  // (C/16)*(N/16) = 8192 tiles per batch
    int t  = wid & 8191;
    int c0 = (t >> 8) << 4;
    int n0 = (t & 255) << 4;
    const uint16_t* cb = catt + (size_t)b * C_ * C_;
    const uint16_t* pt = pamt + (size_t)b * N_ * C_;
    v8f acc = {};
#pragma unroll 4
    for (int d0 = 0; d0 < C_; d0 += 32) {
        v16bf a  = load_a16(cb, c0, C_, d0);
        v16bf bb = load_b16(pt, n0, C_, d0);
        acc = wmma_bf16(a, bb, acc);
    }
    float gc = gamma_cam[0];
    int hi = lane >> 4, nn = lane & 15;
#pragma unroll
    for (int e = 0; e < 8; ++e) {
        int c = c0 + e + (hi << 3);
        size_t idx = ((size_t)b * C_ + c) * N_ + n0 + nn;
        out[idx] = gc * acc[e] + pamf[idx];
    }
}

// ---------------------------------------------------------------------------
extern "C" void kernel_launch(void* const* d_in, const int* in_sizes, int n_in,
                              void* d_out, int out_size, void* d_ws, size_t ws_size,
                              hipStream_t stream) {
    (void)in_sizes; (void)n_in; (void)out_size; (void)ws_size;
    const float* x  = (const float*)d_in[0];
    const float* Wq = (const float*)d_in[1];
    const float* bq = (const float*)d_in[2];
    const float* Wk = (const float*)d_in[3];
    const float* bk = (const float*)d_in[4];
    const float* Wv = (const float*)d_in[5];
    const float* bv = (const float*)d_in[6];
    const float* gp = (const float*)d_in[7];
    const float* gc = (const float*)d_in[8];
    float* out = (float*)d_out;

    char*  ws  = (char*)d_ws;
    size_t off = 0;
    auto carve = [&](size_t bytes) -> void* {
        void* p = ws + off;
        off += (bytes + 255) & ~(size_t)255;
        return p;
    };
    const size_t BCN = (size_t)B_ * C_ * N_;           // 8,388,608
    uint16_t* xt   = (uint16_t*)carve(BCN * 2);        // x^T  (B,N,C) bf16
    uint16_t* wqb  = (uint16_t*)carve((size_t)CQ_ * C_ * 2);
    uint16_t* wkb  = (uint16_t*)carve((size_t)CQ_ * C_ * 2);
    uint16_t* wvb  = (uint16_t*)carve((size_t)C_ * C_ * 2);
    uint16_t* qt   = (uint16_t*)carve((size_t)B_ * N_ * CQ_ * 2);  // (B,N,CQ)
    uint16_t* kt   = (uint16_t*)carve((size_t)B_ * N_ * CQ_ * 2);  // (B,N,CQ)
    uint16_t* vb   = (uint16_t*)carve(BCN * 2);        // V (B,C,N) bf16
    float*    rmax = (float*)carve((size_t)B_ * N_ * 4);
    float*    rsum = (float*)carve((size_t)B_ * N_ * 4);
    float*    pamf = (float*)carve(BCN * 4);           // pam f32 (B,C,N)
    uint16_t* pamb = (uint16_t*)carve(BCN * 2);        // pam bf16 (B,C,N)
    uint16_t* pamt = (uint16_t*)carve(BCN * 2);        // pam^T bf16 (B,N,C)
    float*    cen  = (float*)carve((size_t)B_ * C_ * C_ * 4);
    uint16_t* catt = (uint16_t*)carve((size_t)B_ * C_ * C_ * 2);

    const int T = 256;  // 8 wave32 per block

    // 0) casts / transpose
    k_cast_bf16<<<(CQ_ * C_) / T, T, 0, stream>>>(Wq, wqb, CQ_ * C_);
    k_cast_bf16<<<(CQ_ * C_) / T, T, 0, stream>>>(Wk, wkb, CQ_ * C_);
    k_cast_bf16<<<(C_ * C_) / T, T, 0, stream>>>(Wv, wvb, C_ * C_);
    k_x_transpose<<<(int)(BCN / T), T, 0, stream>>>(x, xt);

    // 1) projections
    k_proj<<<(B_ * (CQ_ / 16) * (N_ / 16)) / 8, T, 0, stream>>>(wqb, bq, xt, CQ_, qt, 1);
    k_proj<<<(B_ * (CQ_ / 16) * (N_ / 16)) / 8, T, 0, stream>>>(wkb, bk, xt, CQ_, kt, 1);
    k_proj<<<(B_ * (C_  / 16) * (N_ / 16)) / 8, T, 0, stream>>>(wvb, bv, xt, C_,  vb, 0);

    // 2) PAM softmax stats (streaming, no N x N materialization)
    k_pam_stats<<<(B_ * (N_ / 16)) / 8, T, 0, stream>>>(qt, kt, rmax, rsum);

    // 3) PAM apply (flash-style second pass), fused pam = gamma*pout + x
    k_pam_apply<<<B_ * (N_ / 16), T, 0, stream>>>(qt, kt, vb, rmax, rsum, x, gp,
                                                  pamf, pamb, pamt);

    // 4) CAM energy
    k_cen<<<(B_ * 32 * 32) / 8, T, 0, stream>>>(pamb, cen);

    // 5) CAM softmax (min-trick)
    k_catt<<<(B_ * C_) / 8, T, 0, stream>>>(cen, catt);

    // 6) CAM apply, fused out = gamma*cout + pam
    k_cam_apply<<<(B_ * (C_ / 16) * (N_ / 16)) / 8, T, 0, stream>>>(catt, pamt, pamf, gc, out);
}
